// SPPGATLayer_20950850470460
// MI455X (gfx1250) — compile-verified, matrix operand order checked by reading, not run on hardware
//
#include <hip/hip_runtime.h>
#include <math.h>

#define NN   20000   // nodes
#define EE   320000  // edges
#define CC   10000   // chars
#define INF_ 768     // input feature dim
#define NH   8       // heads
#define ND   64      // dim per head
#define HDIM 512     // NH*ND
#define SAH  128     // semantic attention hidden

typedef _Float16 v16h __attribute__((ext_vector_type(16)));
typedef _Float16 v8h  __attribute__((ext_vector_type(8)));
typedef float    v8f  __attribute__((ext_vector_type(8)));
typedef _Float16 h2v  __attribute__((ext_vector_type(2)));

// ---------------- WMMA GEMM: proj[M,512] = feat[M,768] @ W[768,512] ----------------
#define BM 128
#define BN 64
#define BK 32
#define LDA 40  // padded LDS stride in halfs (80 B, 16-byte aligned rows)
#define NT (INF_ / BK)  // 24 K-steps

__device__ __forceinline__ v8h pack8(float4 lo, float4 hi) {
  v8h r;
  r[0] = (_Float16)lo.x; r[1] = (_Float16)lo.y;
  r[2] = (_Float16)lo.z; r[3] = (_Float16)lo.w;
  r[4] = (_Float16)hi.x; r[5] = (_Float16)hi.y;
  r[6] = (_Float16)hi.z; r[7] = (_Float16)hi.w;
  return r;
}

__global__ __launch_bounds__(256) void gemm_wmma(const float* __restrict__ feat,
                                                 const float* __restrict__ W,
                                                 float* __restrict__ proj, int M) {
  __shared__ alignas(16) _Float16 As[2][BM * LDA];
  __shared__ alignas(16) _Float16 Bs[2][BN * LDA];
  const int t    = threadIdx.x;
  const int lane = t & 31;
  const int wv   = t >> 5;       // wave 0..7 -> 16-row strip
  const int ml   = lane & 15;
  const int hs   = lane >> 4;
  const int m0   = blockIdx.x * BM;
  const int n0   = blockIdx.y * BN;

  // A staging: thread covers rows arow0 and arow0+64, 8 k-values starting at acol
  const int arow0 = t >> 2;                    // 0..63
  const int acol  = (t & 3) * 8;               // 0,8,16,24
  const int ag0   = min(m0 + arow0, M - 1);    // clamp: branch-free OOB handling
  const int ag1   = min(m0 + arow0 + 64, M - 1);
  // W staging: thread covers k-pair 2*wp, 4 n-values starting at n0+wnb
  const int wp  = t >> 4;                      // 0..15
  const int wnb = (t & 15) * 4;                // 0..60

  float4 a00, a01, a10, a11, w0, w1;

  auto gload = [&](int kt) {
    const int k0 = kt * BK;
    a00 = *(const float4*)&feat[(size_t)ag0 * INF_ + k0 + acol];
    a01 = *(const float4*)&feat[(size_t)ag0 * INF_ + k0 + acol + 4];
    a10 = *(const float4*)&feat[(size_t)ag1 * INF_ + k0 + acol];
    a11 = *(const float4*)&feat[(size_t)ag1 * INF_ + k0 + acol + 4];
    w0  = *(const float4*)&W[(size_t)(k0 + 2 * wp) * HDIM + n0 + wnb];
    w1  = *(const float4*)&W[(size_t)(k0 + 2 * wp + 1) * HDIM + n0 + wnb];
  };
  auto sstore = [&](int b) {
    *(v8h*)&As[b][arow0 * LDA + acol]        = pack8(a00, a01);
    *(v8h*)&As[b][(arow0 + 64) * LDA + acol] = pack8(a10, a11);
    const float* f0 = &w0.x;
    const float* f1 = &w1.x;
#pragma unroll
    for (int j = 0; j < 4; ++j) {
      h2v p;
      p[0] = (_Float16)f0[j];
      p[1] = (_Float16)f1[j];
      *(h2v*)&Bs[b][(wnb + j) * LDA + 2 * wp] = p;
    }
  };

  v8f acc[4] = {};

  gload(0);
  sstore(0);
  __syncthreads();

  for (int kt = 0; kt < NT; ++kt) {
    const int cur = kt & 1;
    if (kt + 1 < NT) gload(kt + 1);
    if (kt + 2 < NT)  // speculative prefetch two slabs ahead (global_prefetch_b8)
      __builtin_prefetch(&feat[(size_t)ag0 * INF_ + (kt + 2) * BK + acol], 0, 1);

    // A fragment: 16-bit A-matrix VGPR striping
    v16h a;
#pragma unroll
    for (int p = 0; p < 8; ++p) {
      int kb = ((p >> 2) << 4) + (hs << 3) + ((p & 3) << 1);
      h2v x = *(const h2v*)&As[cur][(wv * 16 + ml) * LDA + kb];
      a[2 * p] = x[0];
      a[2 * p + 1] = x[1];
    }
#pragma unroll
    for (int nt = 0; nt < 4; ++nt) {
      v16h b;
#pragma unroll
      for (int p = 0; p < 8; ++p) {
        int kb = ((p >> 2) << 4) + (hs << 3) + ((p & 3) << 1);
        h2v x = *(const h2v*)&Bs[cur][(nt * 16 + ml) * LDA + kb];
        b[2 * p] = x[0];
        b[2 * p + 1] = x[1];
      }
      acc[nt] = __builtin_amdgcn_wmma_f32_16x16x32_f16(
          false, a, false, b, (short)0, acc[nt], false, false);
    }

    if (kt + 1 < NT) sstore(cur ^ 1);  // other buffer: no WAR with this iter's reads
    __syncthreads();
  }

  // C/D layout: VGPR v -> row v (lanes 0-15) / v+8 (lanes 16-31), col = ml
#pragma unroll
  for (int nt = 0; nt < 4; ++nt) {
    int col = n0 + nt * 16 + ml;
#pragma unroll
    for (int v = 0; v < 8; ++v) {
      int gm = m0 + wv * 16 + hs * 8 + v;
      if (gm < M) proj[(size_t)gm * HDIM + col] = acc[nt][v];
    }
  }
}

// ---------------- attention terms el/er [N,H] ----------------
__global__ __launch_bounds__(256) void attn_terms_k(const float* __restrict__ proj,
                                                    const float* __restrict__ al,
                                                    const float* __restrict__ ar,
                                                    float* __restrict__ el,
                                                    float* __restrict__ er) {
  int idx = blockIdx.x * blockDim.x + threadIdx.x;
  if (idx >= NN * NH) return;
  int n = idx / NH, h = idx % NH;
  float sl = 0.0f, sr = 0.0f;
  const float* hp  = proj + (size_t)n * HDIM + h * ND;
  const float* alp = al + h * ND;
  const float* arp = ar + h * ND;
#pragma unroll 4
  for (int d = 0; d < ND; ++d) {
    float v = hp[d];
    sl += v * alp[d];
    sr += v * arp[d];
  }
  el[idx] = sl;
  er[idx] = sr;
}

__global__ __launch_bounds__(256) void fill_k(float* p, float v, int n) {
  int i = blockIdx.x * blockDim.x + threadIdx.x;
  if (i < n) p[i] = v;
}

__device__ __forceinline__ void atomicMaxFloat(float* addr, float val) {
  if (!(val == val)) return;
  if (val >= 0.0f)
    atomicMax((int*)addr, __float_as_int(val));
  else
    atomicMin((unsigned int*)addr, (unsigned int)__float_as_int(val));
}

// ---------------- edge pass 1: leaky_relu + segment max ----------------
__global__ __launch_bounds__(256) void edge_max_k(const int* __restrict__ src,
                                                  const int* __restrict__ dst,
                                                  const float* __restrict__ el,
                                                  const float* __restrict__ er,
                                                  float* __restrict__ eat,
                                                  float* __restrict__ mmax) {
  int idx = blockIdx.x * blockDim.x + threadIdx.x;
  if (idx >= EE * NH) return;
  int e = idx / NH, h = idx % NH;
  float x = el[src[e] * NH + h] + er[dst[e] * NH + h];
  x = (x > 0.0f) ? x : 0.2f * x;  // leaky_relu(0.2)
  eat[idx] = x;
  atomicMaxFloat(&mmax[dst[e] * NH + h], x);
}

__global__ __launch_bounds__(256) void max_fixup_k(float* m, int n) {
  int i = blockIdx.x * blockDim.x + threadIdx.x;
  if (i >= n) return;
  float v = m[i];
  if (!isfinite(v)) m[i] = 0.0f;
}

// ---------------- edge pass 2: exp + segment sum ----------------
__global__ __launch_bounds__(256) void edge_exp_k(const int* __restrict__ dst,
                                                  float* __restrict__ eat,
                                                  const float* __restrict__ mmax,
                                                  float* __restrict__ denom) {
  int idx = blockIdx.x * blockDim.x + threadIdx.x;
  if (idx >= EE * NH) return;
  int e = idx / NH, h = idx % NH;
  float v = __expf(eat[idx] - mmax[dst[e] * NH + h]);
  eat[idx] = v;
  atomicAdd(&denom[dst[e] * NH + h], v);
}

// ---------------- edge pass 3: weighted scatter-add (one block per edge) ----------------
__global__ __launch_bounds__(256) void scatter_k(const int* __restrict__ src,
                                                 const int* __restrict__ dst,
                                                 const float* __restrict__ eat,
                                                 const float* __restrict__ denom,
                                                 const float* __restrict__ proj,
                                                 float* __restrict__ aggr) {
  int e = blockIdx.x;
  int s = src[e], d = dst[e];
  const float* hp = proj + (size_t)s * HDIM;
  float* op = aggr + (size_t)d * HDIM;
#pragma unroll
  for (int r = 0; r < 2; ++r) {
    int i = threadIdx.x + r * 256;
    int h = i >> 6;
    float a = eat[e * NH + h] / fmaxf(denom[d * NH + h], 1e-9f);
    atomicAdd(&op[i], a * hp[i]);
  }
}

// ---------------- bias + elu ----------------
__global__ __launch_bounds__(256) void bias_elu_k(float* __restrict__ aggr,
                                                  const float* __restrict__ bias) {
  int idx = blockIdx.x * blockDim.x + threadIdx.x;
  if (idx >= NN * HDIM) return;
  float x = aggr[idx] + bias[idx % HDIM];
  aggr[idx] = (x > 0.0f) ? x : (__expf(x) - 1.0f);
}

// ---------------- semantic attention: per-(c,m) score ----------------
__global__ __launch_bounds__(128) void sem_score_k(const float* __restrict__ z0,
                                                   const int* __restrict__ i0,
                                                   const float* __restrict__ z1,
                                                   const int* __restrict__ i1,
                                                   const float* __restrict__ w1,
                                                   const float* __restrict__ b1,
                                                   const float* __restrict__ w2,
                                                   float* __restrict__ scores) {
  int c = blockIdx.x, m = blockIdx.y, j = threadIdx.x;
  const float* zsrc = (m == 0) ? z0 : z1;
  const int* idx = (m == 0) ? i0 : i1;
  int row = idx ? idx[c] : c;
  const float* z = zsrc + (size_t)row * HDIM;
  float s = b1[j];
  for (int k = 0; k < HDIM; ++k) s += z[k] * w1[k * SAH + j];
  float v = tanhf(s) * w2[j];
  __shared__ float red[128];
  red[j] = v;
  __syncthreads();
  for (int st = 64; st > 0; st >>= 1) {
    if (j < st) red[j] += red[j + st];
    __syncthreads();
  }
  if (j == 0) atomicAdd(&scores[m], red[0]);
}

__global__ void beta_k(float* s) {
  float s0 = s[0] / (float)CC, s1 = s[1] / (float)CC;
  float mx = fmaxf(s0, s1);
  float e0 = __expf(s0 - mx), e1 = __expf(s1 - mx);
  float d = e0 + e1;
  s[2] = e0 / d;
  s[3] = e1 / d;
}

__global__ __launch_bounds__(256) void sem_combine_k(const float* __restrict__ z0,
                                                     const int* __restrict__ i0,
                                                     const float* __restrict__ z1,
                                                     const int* __restrict__ i1,
                                                     const float* __restrict__ beta,
                                                     float* __restrict__ out) {
  int idx = blockIdx.x * blockDim.x + threadIdx.x;
  if (idx >= CC * HDIM) return;
  int c = idx / HDIM, i = idx % HDIM;
  int r0 = i0 ? i0[c] : c;
  int r1 = i1 ? i1[c] : c;
  out[idx] = beta[2] * z0[(size_t)r0 * HDIM + i] + beta[3] * z1[(size_t)r1 * HDIM + i];
}

// ---------------- head means and char override ----------------
__global__ __launch_bounds__(256) void mean_k(const float* __restrict__ aggr,
                                              float* __restrict__ out) {
  int idx = blockIdx.x * blockDim.x + threadIdx.x;
  if (idx >= NN * ND) return;
  int n = idx / ND, d = idx % ND;
  float s = 0.0f;
#pragma unroll
  for (int h = 0; h < NH; ++h) s += aggr[(size_t)n * HDIM + h * ND + d];
  out[idx] = s * 0.125f;
}

__global__ __launch_bounds__(256) void override_k(const float* __restrict__ sem,
                                                  const int* __restrict__ chr,
                                                  float* __restrict__ outmean) {
  int idx = blockIdx.x * blockDim.x + threadIdx.x;
  if (idx >= CC * ND) return;
  int c = idx / ND, d = idx % ND;
  float s = 0.0f;
#pragma unroll
  for (int h = 0; h < NH; ++h) s += sem[(size_t)c * HDIM + h * ND + d];
  outmean[(size_t)chr[c] * ND + d] = s * 0.125f;
}

// ---------------- host launcher ----------------
extern "C" void kernel_launch(void* const* d_in, const int* in_sizes, int n_in,
                              void* d_out, int out_size, void* d_ws, size_t ws_size,
                              hipStream_t stream) {
  (void)in_sizes; (void)n_in; (void)out_size; (void)ws_size;
  const float* feat[3] = {(const float*)d_in[0], (const float*)d_in[8], (const float*)d_in[16]};
  const int*   srcA[3] = {(const int*)d_in[1], (const int*)d_in[9], (const int*)d_in[17]};
  const int*   dstA[3] = {(const int*)d_in[2], (const int*)d_in[10], (const int*)d_in[18]};
  const int*   chr[3]  = {(const int*)d_in[3], (const int*)d_in[11], (const int*)d_in[19]};
  const float* Wm[3]   = {(const float*)d_in[4], (const float*)d_in[12], (const float*)d_in[20]};
  const float* al[3]   = {(const float*)d_in[5], (const float*)d_in[13], (const float*)d_in[21]};
  const float* ar[3]   = {(const float*)d_in[6], (const float*)d_in[14], (const float*)d_in[22]};
  const float* bias[3] = {(const float*)d_in[7], (const float*)d_in[15], (const float*)d_in[23]};
  const float* sw1 = (const float*)d_in[24];
  const float* sb1 = (const float*)d_in[25];
  const float* sw2 = (const float*)d_in[26];
  const float* aw1 = (const float*)d_in[27];
  const float* ab1 = (const float*)d_in[28];
  const float* aw2 = (const float*)d_in[29];

  float* ws = (float*)d_ws;
  const size_t PROJ = (size_t)NN * HDIM;
  float* proj[3];
  float* aggr[3];
  for (int i = 0; i < 3; ++i) { proj[i] = ws + i * PROJ; aggr[i] = ws + (3 + i) * PROJ; }
  float* el     = ws + 6 * PROJ;
  float* er     = el + (size_t)NN * NH;
  float* eat    = er + (size_t)NN * NH;
  float* mmax   = eat + (size_t)EE * NH;
  float* denom  = mmax + (size_t)NN * NH;
  float* sound  = denom + (size_t)NN * NH;
  float* scores = sound + (size_t)CC * HDIM;

  float* out     = (float*)d_out;
  float* sem_emb = out;                          // [C,512]
  float* h1m     = sem_emb + (size_t)CC * HDIM;  // [N,64]
  float* h2m     = h1m + (size_t)NN * ND;
  float* hhm     = h2m + (size_t)NN * ND;

  const dim3 ggrid((NN + BM - 1) / BM, HDIM / BN);
  const int NHb = (NN * NH + 255) / 256;
  const int EHb = (EE * NH + 255) / 256;

  for (int L = 0; L < 3; ++L) {
    gemm_wmma<<<ggrid, 256, 0, stream>>>(feat[L], Wm[L], proj[L], NN);
    attn_terms_k<<<NHb, 256, 0, stream>>>(proj[L], al[L], ar[L], el, er);
    fill_k<<<NHb, 256, 0, stream>>>(mmax, -INFINITY, NN * NH);
    hipMemsetAsync(denom, 0, (size_t)NN * NH * sizeof(float), stream);
    hipMemsetAsync(aggr[L], 0, PROJ * sizeof(float), stream);
    edge_max_k<<<EHb, 256, 0, stream>>>(srcA[L], dstA[L], el, er, eat, mmax);
    max_fixup_k<<<NHb, 256, 0, stream>>>(mmax, NN * NH);
    edge_exp_k<<<EHb, 256, 0, stream>>>(dstA[L], eat, mmax, denom);
    scatter_k<<<EE, 256, 0, stream>>>(srcA[L], dstA[L], eat, denom, proj[L], aggr[L]);
    bias_elu_k<<<(NN * HDIM + 255) / 256, 256, 0, stream>>>(aggr[L], bias[L]);
  }

  // semantic attention 1: z = [h1[char0], hh[char2]] -> sound
  hipMemsetAsync(scores, 0, 4 * sizeof(float), stream);
  dim3 sgrid(CC, 2);
  sem_score_k<<<sgrid, 128, 0, stream>>>(aggr[0], chr[0], aggr[2], chr[2], sw1, sb1, sw2, scores);
  beta_k<<<1, 1, 0, stream>>>(scores);
  sem_combine_k<<<(CC * HDIM + 255) / 256, 256, 0, stream>>>(aggr[0], chr[0], aggr[2], chr[2],
                                                             scores, sound);
  // semantic attention 2: z2 = [sound, h2[char1]] -> sem_emb
  hipMemsetAsync(scores, 0, 4 * sizeof(float), stream);
  sem_score_k<<<sgrid, 128, 0, stream>>>(sound, nullptr, aggr[1], chr[1], aw1, ab1, aw2, scores);
  beta_k<<<1, 1, 0, stream>>>(scores);
  sem_combine_k<<<(CC * HDIM + 255) / 256, 256, 0, stream>>>(sound, nullptr, aggr[1], chr[1],
                                                             scores, sem_emb);

  const int NDb = (NN * ND + 255) / 256;
  mean_k<<<NDb, 256, 0, stream>>>(aggr[0], h1m);
  mean_k<<<NDb, 256, 0, stream>>>(aggr[1], h2m);
  mean_k<<<NDb, 256, 0, stream>>>(aggr[2], hhm);
  const int CDb = (CC * ND + 255) / 256;
  override_k<<<CDb, 256, 0, stream>>>(sem_emb, chr[0], h1m);
  override_k<<<CDb, 256, 0, stream>>>(sem_emb, chr[1], h2m);
  override_k<<<CDb, 256, 0, stream>>>(sem_emb, chr[2], hhm);
}
